// TransfomerBlock_44032004719198
// MI455X (gfx1250) — compile-verified, matrix operand order checked by reading, not run on hardware
//
#include <hip/hip_runtime.h>
#include <hip/hip_bf16.h>
#include <math.h>

typedef __attribute__((ext_vector_type(16))) __bf16 v16bf;
typedef __attribute__((ext_vector_type(8)))  __bf16 v8bf;
typedef __attribute__((ext_vector_type(8)))  float  v8f;

// ---------------------------------------------------------------------------
// CDNA5 helpers
// ---------------------------------------------------------------------------
__device__ __forceinline__ v8f wmma_bf16(v16bf a, v16bf b, v8f c) {
  return __builtin_amdgcn_wmma_f32_16x16x32_bf16(
      /*neg_a=*/false, a, /*neg_b=*/false, b,
      /*c_mod=*/(short)0, c, /*reuse_a=*/false, /*reuse_b=*/false);
}

// Async DMA: 16 bytes per lane, global -> LDS, tracked by ASYNCcnt.
// LDS dest address = low 32 bits of the flat shared pointer.
__device__ __forceinline__ void async_cp_b128(void* lds_dst, const void* gsrc) {
  unsigned lds_off = (unsigned)(unsigned long long)lds_dst;
  asm volatile("global_load_async_to_lds_b128 %0, %1, off"
               :: "v"(lds_off), "v"(gsrc)
               : "memory");
}

__device__ __forceinline__ void wait_async0() {
  asm volatile("s_wait_asynccnt 0x0" ::: "memory");
}

// Load a 16x32 bf16 A/B fragment from a row-major [rows][ld] bf16 matrix in
// global memory.  Lane layout (ISA 7.12.2): lane&15 = row, lane>>4 = K-half;
// elems 0..7 = K 8h..8h+7, elems 8..15 = K 16+8h..16+8h+7 (two b128 loads).
__device__ __forceinline__ v16bf load_frag(const __bf16* __restrict__ base,
                                           int ld, int row, int col) {
  const int lane = threadIdx.x & 31;
  const int rc = lane & 15, half = lane >> 4;
  const __bf16* p = base + (long)(row + rc) * ld + col + 8 * half;
  v8bf lo = *(const v8bf*)p;
  v8bf hi = *(const v8bf*)(p + 16);
  return __builtin_shufflevector(lo, hi, 0, 1, 2, 3, 4, 5, 6, 7,
                                 8, 9, 10, 11, 12, 13, 14, 15);
}

// Same fragment layout but from an LDS row pointer (row = lane&15 already
// applied by the caller; the row holds a 32-wide K window).
__device__ __forceinline__ v16bf load_frag_lds(const __bf16* rowp) {
  const int half = (threadIdx.x >> 4) & 1;
  v8bf lo = *(const v8bf*)(rowp + 8 * half);
  v8bf hi = *(const v8bf*)(rowp + 16 + 8 * half);
  return __builtin_shufflevector(lo, hi, 0, 1, 2, 3, 4, 5, 6, 7,
                                 8, 9, 10, 11, 12, 13, 14, 15);
}

// ---------------------------------------------------------------------------
// fp32 -> bf16 conversion / weight transpose / V transpose
// ---------------------------------------------------------------------------
__global__ void f32_to_bf16_kernel(const float* __restrict__ in,
                                   __bf16* __restrict__ out, long n) {
  long i = (long)blockIdx.x * blockDim.x + threadIdx.x;
  if (i < n) out[i] = (__bf16)in[i];
}

// W[K][N] fp32 -> Wt[N][K] bf16 (weight columns become contiguous B rows)
__global__ void transpose_to_bf16_kernel(const float* __restrict__ W,
                                         __bf16* __restrict__ Wt,
                                         int K, int N) {
  long i = (long)blockIdx.x * blockDim.x + threadIdx.x;
  if (i < (long)K * N) {
    int k = (int)(i / N);
    int n = (int)(i % N);
    Wt[(long)n * K + k] = (__bf16)W[i];
  }
}

// vt[((b*H+h)*64 + d)*T + t] = qkv_bf[(b*T+t)*2304 + 1536 + h*64 + d]
__global__ void build_vt_kernel(const __bf16* __restrict__ qkv,
                                __bf16* __restrict__ vt) {
  const long total = 2L * 12 * 64 * 2048;
  long i = (long)blockIdx.x * blockDim.x + threadIdx.x;
  if (i < total) {
    int t = (int)(i & 2047);
    long r = i >> 11;
    int d = (int)(r & 63); r >>= 6;
    int h = (int)(r % 12);
    int b = (int)(r / 12);
    vt[i] = qkv[((long)(b * 2048 + t)) * 2304 + 1536 + h * 64 + d];
  }
}

// ---------------------------------------------------------------------------
// Generic bf16 WMMA GEMM: out[M][N] = A[M][K] @ Bt[N][K]^T + bias (+GELU)
// block = 256 threads = 8 waves; wave tile = 32(M) x 64(N); K step = 32.
// The 64x32 B tile (shared by all 8 waves) is staged via double-buffered
// async global->LDS DMA; each B fragment feeds two WMMAs (8 wmma / 4 ds-frag
// loads per K step); A fragments are register-pipelined.
// ---------------------------------------------------------------------------
__global__ void __launch_bounds__(256)
gemm_bf16_wmma_kernel(const __bf16* __restrict__ A,
                      const __bf16* __restrict__ Bt,
                      const float* __restrict__ bias,
                      float* __restrict__ outF,
                      __bf16* __restrict__ outB,
                      int M, int N, int K, int doGelu) {
  const int tid = threadIdx.x;
  const int lane = tid & 31, wave = tid >> 5;
  const int rc = lane & 15, half = lane >> 4;
  const int m0 = blockIdx.y * 256 + wave * 32;
  const int n0 = blockIdx.x * 64;

  __shared__ alignas(16) __bf16 Btile[2][64][32];  // [buf][n][k], 2 x 4KB
  const int bn = tid >> 2;            // 0..63   (N row of the tile)
  const int bc = (tid & 3) * 8;       // 0/8/16/24 (K chunk)
  const __bf16* bsrc = Bt + (long)(n0 + bn) * K + bc;

  async_cp_b128(&Btile[0][bn][bc], bsrc);           // stage tile 0

  const int nt = K / 32;
  v8f acc[2][4] = {};
  v16bf af0 = load_frag(A, K, m0, 0);               // A frags, pipelined
  v16bf af1 = load_frag(A, K, m0 + 16, 0);

  for (int i = 0; i < nt; ++i) {
    wait_async0();                                  // my tile-i chunk done
    __syncthreads();                                // everyone's chunks done
    if (i + 1 < nt)                                 // overlap copy(i+1)
      async_cp_b128(&Btile[(i + 1) & 1][bn][bc], bsrc + (long)(i + 1) * 32);
    v16bf a0n = af0, a1n = af1;
    if (i + 1 < nt) {
      a0n = load_frag(A, K, m0, (i + 1) * 32);
      a1n = load_frag(A, K, m0 + 16, (i + 1) * 32);
    }

    const int buf = i & 1;
    // batch all four B fragments so one DS wait covers them
    v16bf b0 = load_frag_lds(&Btile[buf][rc][0]);
    v16bf b1 = load_frag_lds(&Btile[buf][16 + rc][0]);
    v16bf b2 = load_frag_lds(&Btile[buf][32 + rc][0]);
    v16bf b3 = load_frag_lds(&Btile[buf][48 + rc][0]);

    acc[0][0] = wmma_bf16(af0, b0, acc[0][0]);
    acc[1][0] = wmma_bf16(af1, b0, acc[1][0]);
    acc[0][1] = wmma_bf16(af0, b1, acc[0][1]);
    acc[1][1] = wmma_bf16(af1, b1, acc[1][1]);
    acc[0][2] = wmma_bf16(af0, b2, acc[0][2]);
    acc[1][2] = wmma_bf16(af1, b2, acc[1][2]);
    acc[0][3] = wmma_bf16(af0, b3, acc[0][3]);
    acc[1][3] = wmma_bf16(af1, b3, acc[1][3]);

    af0 = a0n; af1 = a1n;
  }

#pragma unroll
  for (int t = 0; t < 2; ++t) {
#pragma unroll
    for (int j = 0; j < 4; ++j) {
      const int n = n0 + 16 * j + rc;
      const float bv = bias ? bias[n] : 0.0f;
#pragma unroll
      for (int r = 0; r < 8; ++r) {
        const int m = m0 + 16 * t + r + 8 * half;
        float v = acc[t][j][r] + bv;
        if (doGelu) v = 0.5f * v * (1.0f + erff(v * 0.70710678118654752f));
        const long idx = (long)m * N + n;
        if (outF) outF[idx] = v;
        if (outB) outB[idx] = (__bf16)v;
      }
    }
  }
}

// ---------------------------------------------------------------------------
// Flash attention: per wave 16 query rows, streaming key blocks of 32.
// K and V^T blocks are identical across the 8 waves -> staged through
// double-buffered LDS with async DMA.  S = Q Kt (4 wmma), online softmax
// (shfl_xor row reductions within the 16-lane halves), P staged via
// per-wave LDS, O += P V (4 wmma).
// ---------------------------------------------------------------------------
__global__ void __launch_bounds__(256)
flash_attn_wmma_kernel(const __bf16* __restrict__ qkv,
                       const __bf16* __restrict__ vt,
                       const int* __restrict__ mask,
                       __bf16* __restrict__ attnB) {
  constexpr int T = 2048, C3 = 2304, C = 768, DH = 64, H = 12;
  const int tid = threadIdx.x;
  const int lane = tid & 31, wave = tid >> 5;
  const int rc = lane & 15, half = lane >> 4;
  const int h = blockIdx.y, b = blockIdx.z;
  const int q0 = blockIdx.x * 128 + wave * 16;

  const __bf16* qbase = qkv + (long)b * T * C3 + h * DH;   // rows = query t
  const __bf16* kbase = qbase + C;                         // rows = key t
  const __bf16* vtbase = vt + (long)(b * H + h) * DH * T;  // rows = d, cols = t

  __shared__ alignas(16) __bf16 Ktile[2][32][64];  // [buf][key][d]   2 x 4KB
  __shared__ alignas(16) __bf16 Vtile[2][64][32];  // [buf][d][key]   2 x 4KB
  __shared__ alignas(16) __bf16 plds[8][16][32];   // per-wave P tile    8KB

  const int kr = tid >> 3, kc = (tid & 7) * 8;     // K stage: key row, d chunk
  const int vr = tid >> 2, vc = (tid & 3) * 8;     // V stage: d row, key chunk

  // stage key-block 0
  async_cp_b128(&Ktile[0][kr][kc], kbase + (long)kr * C3 + kc);
  async_cp_b128(&Vtile[0][vr][vc], vtbase + (long)vr * T + vc);

  const v16bf qf0 = load_frag(qbase, C3, q0, 0);
  const v16bf qf1 = load_frag(qbase, C3, q0, 32);

  v8f o0 = {}, o1 = {}, o2 = {}, o3 = {};
  float mrow[8], lrow[8];
#pragma unroll
  for (int r = 0; r < 8; ++r) { mrow[r] = -1e30f; lrow[r] = 0.0f; }

  constexpr int NBLK = T / 32;
  for (int i = 0; i < NBLK; ++i) {
    const int kb = i * 32;
    wait_async0();
    __syncthreads();
    if (i + 1 < NBLK) {
      async_cp_b128(&Ktile[(i + 1) & 1][kr][kc],
                    kbase + (long)(kb + 32 + kr) * C3 + kc);
      async_cp_b128(&Vtile[(i + 1) & 1][vr][vc],
                    vtbase + (long)vr * T + kb + 32 + vc);
    }
    const int buf = i & 1;

    // ---- S = (Q Kt) for 32 keys (two 16x16 tiles); batch the DS loads ----
    v16bf k00 = load_frag_lds(&Ktile[buf][rc][0]);
    v16bf k01 = load_frag_lds(&Ktile[buf][rc][32]);
    v16bf k10 = load_frag_lds(&Ktile[buf][16 + rc][0]);
    v16bf k11 = load_frag_lds(&Ktile[buf][16 + rc][32]);
    v8f s0 = {}, s1 = {};
    s0 = wmma_bf16(qf0, k00, s0);
    s0 = wmma_bf16(qf1, k01, s0);
    s1 = wmma_bf16(qf0, k10, s1);
    s1 = wmma_bf16(qf1, k11, s1);

    const bool ok0 = mask[(long)b * T + kb + rc] != 0;
    const bool ok1 = mask[(long)b * T + kb + 16 + rc] != 0;

    // ---- online softmax over the key axis (lanes within each half) ----
#pragma unroll
    for (int r = 0; r < 8; ++r) {
      float a0 = ok0 ? s0[r] * 0.125f : -1e30f;
      float a1 = ok1 ? s1[r] * 0.125f : -1e30f;
      float mx = fmaxf(a0, a1);
#pragma unroll
      for (int off = 1; off < 16; off <<= 1)
        mx = fmaxf(mx, __shfl_xor(mx, off, 32));
      const float mnew = fmaxf(mrow[r], mx);
      const float alpha = __expf(mrow[r] - mnew);
      mrow[r] = mnew;
      const float p0 = __expf(a0 - mnew);
      const float p1 = __expf(a1 - mnew);
      float ps = p0 + p1;
#pragma unroll
      for (int off = 1; off < 16; off <<= 1)
        ps += __shfl_xor(ps, off, 32);
      lrow[r] = lrow[r] * alpha + ps;
      o0[r] *= alpha; o1[r] *= alpha; o2[r] *= alpha; o3[r] *= alpha;
      // D-layout (m = r + 8*half, n = rc) -> LDS row-major P tile
      plds[wave][r + 8 * half][rc]      = (__bf16)p0;
      plds[wave][r + 8 * half][16 + rc] = (__bf16)p1;
    }

    // ---- reload P in A-fragment layout (same wave; DScnt ordering) ----
    const __bf16* lp = &plds[wave][rc][0];
    v8bf plo = *(const v8bf*)(lp + 8 * half);
    v8bf phi = *(const v8bf*)(lp + 16 + 8 * half);
    v16bf pf = __builtin_shufflevector(plo, phi, 0, 1, 2, 3, 4, 5, 6, 7,
                                       8, 9, 10, 11, 12, 13, 14, 15);

    // ---- O += P V  (Vtile rows are head-dims, contiguous over keys) ----
    v16bf vb0 = load_frag_lds(&Vtile[buf][rc][0]);
    v16bf vb1 = load_frag_lds(&Vtile[buf][16 + rc][0]);
    v16bf vb2 = load_frag_lds(&Vtile[buf][32 + rc][0]);
    v16bf vb3 = load_frag_lds(&Vtile[buf][48 + rc][0]);
    o0 = wmma_bf16(pf, vb0, o0);
    o1 = wmma_bf16(pf, vb1, o1);
    o2 = wmma_bf16(pf, vb2, o2);
    o3 = wmma_bf16(pf, vb3, o3);
  }

  // ---- normalize and scatter into attn (B,T,C) as bf16 ----
#pragma unroll
  for (int r = 0; r < 8; ++r) {
    const float inv = 1.0f / lrow[r];
    const long m = q0 + r + 8 * half;
    __bf16* dst = attnB + ((long)b * T + m) * C + h * DH;
    dst[rc]      = (__bf16)(o0[r] * inv);
    dst[16 + rc] = (__bf16)(o1[r] * inv);
    dst[32 + rc] = (__bf16)(o2[r] * inv);
    dst[48 + rc] = (__bf16)(o3[r] * inv);
  }
}

// ---------------------------------------------------------------------------
// out = resid + LayerNorm(src) * gamma + beta   (row-wise, biased variance)
// one block (256 threads) per row of C=768.
// ---------------------------------------------------------------------------
__global__ void __launch_bounds__(256)
ln_residual_kernel(const float* __restrict__ src,
                   const float* __restrict__ resid,
                   const float* __restrict__ gamma,
                   const float* __restrict__ beta,
                   float* __restrict__ outF,
                   __bf16* __restrict__ outB, int C) {
  const long row = blockIdx.x;
  const float* s = src + row * C;
  const float* rz = resid + row * C;
  __shared__ float rsum[256], rsq[256];
  const int tid = threadIdx.x;

  float sum = 0.0f, sq = 0.0f;
  for (int c = tid; c < C; c += 256) {
    float v = s[c];
    sum += v; sq += v * v;
  }
  rsum[tid] = sum; rsq[tid] = sq;
  __syncthreads();
  for (int off = 128; off > 0; off >>= 1) {
    if (tid < off) { rsum[tid] += rsum[tid + off]; rsq[tid] += rsq[tid + off]; }
    __syncthreads();
  }
  const float mean = rsum[0] / (float)C;
  const float var  = rsq[0] / (float)C - mean * mean;
  const float inv  = rsqrtf(var + 1e-5f);

  for (int c = tid; c < C; c += 256) {
    float v = rz[c] + (s[c] - mean) * inv * gamma[c] + beta[c];
    if (outF) outF[row * C + c] = v;
    if (outB) outB[row * C + c] = (__bf16)v;
  }
}

// ---------------------------------------------------------------------------
// Launcher
// ---------------------------------------------------------------------------
extern "C" void kernel_launch(void* const* d_in, const int* in_sizes, int n_in,
                              void* d_out, int out_size, void* d_ws, size_t ws_size,
                              hipStream_t stream) {
  (void)in_sizes; (void)n_in; (void)out_size; (void)ws_size;
  constexpr int B = 2, T = 2048, C = 768, H = 12, Dff = 3072;
  constexpr int M = B * T;          // 4096 rows
  constexpr int C3 = 3 * C;         // 2304

  const float* x      = (const float*)d_in[0];
  const int*   mask   = (const int*)d_in[1];
  const float* Wqkv   = (const float*)d_in[2];
  const float* bqkv   = (const float*)d_in[3];
  const float* Wp     = (const float*)d_in[4];
  const float* bp     = (const float*)d_in[5];
  const float* gamma1 = (const float*)d_in[6];
  const float* beta1  = (const float*)d_in[7];
  const float* W1     = (const float*)d_in[8];
  const float* b1     = (const float*)d_in[9];
  const float* W2     = (const float*)d_in[10];
  const float* b2     = (const float*)d_in[11];
  const float* gamma2 = (const float*)d_in[12];
  const float* beta2  = (const float*)d_in[13];
  float* out = (float*)d_out;

  char* p = (char*)d_ws;
  auto alloc = [&](size_t bytes) -> void* {
    void* r = (void*)p;
    p += (bytes + 255) & ~(size_t)255;
    return r;
  };
  __bf16* Xbf    = (__bf16*)alloc((size_t)M * C * 2);
  __bf16* Wqkvt  = (__bf16*)alloc((size_t)C3 * C * 2);
  __bf16* Wpt    = (__bf16*)alloc((size_t)C * C * 2);
  __bf16* W1t    = (__bf16*)alloc((size_t)Dff * C * 2);
  __bf16* W2t    = (__bf16*)alloc((size_t)C * Dff * 2);
  __bf16* qkvB   = (__bf16*)alloc((size_t)M * C3 * 2);
  __bf16* Vt     = (__bf16*)alloc((size_t)B * H * 64 * T * 2);
  __bf16* attnB  = (__bf16*)alloc((size_t)M * C * 2);
  float*  projF  = (float*)alloc((size_t)M * C * 4);
  float*  out1F  = (float*)alloc((size_t)M * C * 4);
  __bf16* out1B  = (__bf16*)alloc((size_t)M * C * 2);
  __bf16* hB     = (__bf16*)alloc((size_t)M * Dff * 2);
  float*  mlpF   = (float*)alloc((size_t)M * C * 4);

  const dim3 blk(256);

  // 1. activations + weights -> bf16 (weights transposed to [N][K])
  {
    long n = (long)M * C;
    f32_to_bf16_kernel<<<dim3((n + 255) / 256), blk, 0, stream>>>(x, Xbf, n);
    long nw = (long)C * C3;
    transpose_to_bf16_kernel<<<dim3((nw + 255) / 256), blk, 0, stream>>>(Wqkv, Wqkvt, C, C3);
    nw = (long)C * C;
    transpose_to_bf16_kernel<<<dim3((nw + 255) / 256), blk, 0, stream>>>(Wp, Wpt, C, C);
    nw = (long)C * Dff;
    transpose_to_bf16_kernel<<<dim3((nw + 255) / 256), blk, 0, stream>>>(W1, W1t, C, Dff);
    nw = (long)Dff * C;
    transpose_to_bf16_kernel<<<dim3((nw + 255) / 256), blk, 0, stream>>>(W2, W2t, Dff, C);
  }

  // 2. QKV = X @ Wqkv + bqkv  (bf16 out only)
  gemm_bf16_wmma_kernel<<<dim3(C3 / 64, M / 256), blk, 0, stream>>>(
      Xbf, Wqkvt, bqkv, nullptr, qkvB, M, C3, C, 0);

  // 3. V transpose for the PV WMMA B-operand
  {
    long n = (long)B * H * 64 * T;
    build_vt_kernel<<<dim3((n + 255) / 256), blk, 0, stream>>>(qkvB, Vt);
  }

  // 4. flash attention -> attn (bf16, (B,T,C))
  flash_attn_wmma_kernel<<<dim3(T / 128, H, B), blk, 0, stream>>>(
      qkvB, Vt, mask, attnB);

  // 5. proj = attn @ Wp + bp (fp32)
  gemm_bf16_wmma_kernel<<<dim3(C / 64, M / 256), blk, 0, stream>>>(
      attnB, Wpt, bp, projF, nullptr, M, C, C, 0);

  // 6. out1 = x + LN(proj)
  ln_residual_kernel<<<dim3(M), blk, 0, stream>>>(
      projF, x, gamma1, beta1, out1F, out1B, C);

  // 7. h = gelu(out1 @ W1 + b1)  (bf16 out only)
  gemm_bf16_wmma_kernel<<<dim3(Dff / 64, M / 256), blk, 0, stream>>>(
      out1B, W1t, b1, nullptr, hB, M, Dff, C, 1);

  // 8. mlp = h @ W2 + b2 (fp32)
  gemm_bf16_wmma_kernel<<<dim3(C / 64, M / 256), blk, 0, stream>>>(
      hB, W2t, b2, mlpF, nullptr, M, C, Dff, 0);

  // 9. out = out1 + LN(mlp)
  ln_residual_kernel<<<dim3(M), blk, 0, stream>>>(
      mlpF, out1F, gamma2, beta2, out, nullptr, C);
}
